// MxAssemble_41042707480843
// MI455X (gfx1250) — matrix operand order repaired and checked by reading, not compiled, not run on hardware
//
#include <hip/hip_runtime.h>
#include <hip/hip_bf16.h>

// MxAssemble for MI455X (gfx1250): banded-Toeplitz GEMM on v_wmma_f32_16x16x32_bf16.
//
// out[b,c,y,x] = sum_{dy,dx in [0,13)} aff[b,dy*13+dx,y,x] * in2[b,c,y+dy-6,x+dx-6]
//
// Per (y, x-tile of 16, dy): out[C,16] += A[C,32] x Bband[32,16], with the K=32
// slab covering source cols u = x0-8 .. x0+23 and
// Bband[t,j] = aff[b, dy*13 + (t-j-2), y, x0+j] for 2 <= t-j <= 14, else 0.
//
// v2: block 4 y-rows x 32 x-cols per 8-wave WG so the in2 halo is staged 6x
// instead of 26x (A patch 16 rows x 48 cols staged once per WG); aff converted
// to bf16 at load so the B band-build is a predicated LDS copy. ~155 KB LDS ->
// 2 WGs / 320KB WGP.

typedef __attribute__((ext_vector_type(16))) __bf16 v16bf;
typedef __attribute__((ext_vector_type(8)))  __bf16 v8bf;
typedef __attribute__((ext_vector_type(8)))  float  v8f;

namespace {
constexpr int KD    = 6;               // max displacement
constexpr int WIN   = 2 * KD + 1;      // 13
constexpr int DN    = WIN * WIN;       // 169
constexpr int BATCH = 4;
constexpr int CH    = 64;
constexpr int H     = 192;
constexpr int W     = 192;
constexpr int TX    = 16;              // WMMA N (one x-tile)
constexpr int XB    = 32;              // x-cols per WG (2 x-tiles)
constexpr int YB    = 4;               // y-rows per WG
constexpr int AROWS = YB + 2 * KD;     // 16 staged in2 rows
constexpr int UW    = XB + 16;         // 48 staged u-cols (u0 = xg-8)
constexpr int TSTR  = 56;              // A t-stride (112B = 28 banks, conflict-free b128)
constexpr int BSTR  = 40;              // B t-stride (80B = 20 banks, conflict-free b128)
constexpr int NTHR  = 256;             // 8 waves
}

__device__ __forceinline__ __bf16 f2bf(float f) {
  // round-to-nearest-even f32 -> bf16
  union { float f; unsigned u; } x; x.f = f;
  unsigned r = x.u + 0x7FFFu + ((x.u >> 16) & 1u);
  unsigned short s = (unsigned short)(r >> 16);
  __bf16 b; __builtin_memcpy(&b, &s, 2);
  return b;
}

__global__ __launch_bounds__(NTHR) void mx_assemble_wmma(
    const float* __restrict__ aff,   // [B, DN, H, W]
    const float* __restrict__ in2,   // [B, CH, H, W]
    float* __restrict__ out)         // [B, CH, H, W]
{
  const int xg  = blockIdx.x * XB;   // 0,32,..,160
  const int y0  = blockIdx.y * YB;   // 0,4,..,188
  const int b   = blockIdx.z;        // 0..3
  const int tid = threadIdx.x;       // 0..255

  // A: in2 patch  [row(16)][c(64)][t(48 of 56)] bf16          -> 114,688 B
  // B: banded aff [dy(13)][xt(2)][j(16)][t(32 of 40)] bf16    ->  33,280 B
  // aff_s: one y-row of aff cols [d(169)][x(32)] bf16         ->  10,816 B
  __shared__ __align__(16) __bf16 A_lds[AROWS * CH * TSTR];
  __shared__ __align__(16) __bf16 B_lds[WIN * 2 * TX * BSTR];
  __shared__ __align__(16) __bf16 aff_s[DN * XB];

  // ---- Stage in2 patch once: rows y0-6..y0+9, cols xg-8..xg+39, zero OOB.
  // Consecutive tids cover a 48-elem (192B) contiguous global run per (row,c).
  for (int idx = tid; idx < AROWS * CH * UW; idx += NTHR) {
    const int t   = idx % UW;
    const int rc  = idx / UW;
    const int c   = rc & (CH - 1);
    const int row = rc >> 6;                 // 0..15
    const int yy  = y0 + row - KD;
    const int xx  = xg - 8 + t;
    float v = 0.0f;
    if ((unsigned)yy < (unsigned)H && (unsigned)xx < (unsigned)W)
      v = in2[(((size_t)b * CH + c) * H + yy) * W + xx];
    A_lds[(row * CH + c) * TSTR + t] = f2bf(v);
  }

  const int wave  = tid >> 5;        // 0..7
  const int lane  = tid & 31;
  const int ctile = wave & 3;        // channel tile (M)
  const int xt    = wave >> 2;       // x-tile within block (N)
  const int mh    = lane >> 4;       // lane half -> K sub-range
  const int ml    = lane & 15;       // A: M row / B: N column

  for (int r = 0; r < YB; ++r) {
    const int y = y0 + r;

    // ---- Stage aff columns for this row: aff_s[d][x] = bf16(aff[b,d,y,xg+x]).
    // 32 consecutive floats (128B) per d -> fully coalesced; each aff element
    // is read from HBM exactly once across the whole grid.
    __syncthreads();   // previous row's consumers done with aff_s / B_lds
    for (int idx = tid; idx < DN * XB; idx += NTHR) {
      const int x = idx & (XB - 1);
      const int d = idx >> 5;
      aff_s[idx] = f2bf(aff[(((size_t)b * DN + d) * H + y) * W + (xg + x)]);
    }
    __syncthreads();

    // ---- Build banded B tiles (pure predicated bf16 copy from LDS).
    for (int idx = tid; idx < WIN * 2 * TX * 32; idx += NTHR) {
      const int t  = idx & 31;
      const int r1 = idx >> 5;
      const int j  = r1 & (TX - 1);
      const int r2 = r1 >> 4;
      const int bxt = r2 & 1;
      const int dy  = r2 >> 1;
      const int dx  = t - j - 2;
      __bf16 v = f2bf(0.0f);
      if ((unsigned)dx < (unsigned)WIN)
        v = aff_s[(dy * WIN + dx) * XB + (bxt * TX + j)];
      B_lds[((dy * 2 + bxt) * TX + j) * BSTR + t] = v;
    }
    __syncthreads();

    // ---- 13 WMMAs per wave for this output row's (C-tile, x-tile).
    v8f acc = {};
#pragma unroll
    for (int dy = 0; dy < WIN; ++dy) {
      // A fragment (16-bit 16x32): K runs [8mh, 8mh+7] and [16+8mh, +7],
      // offset by 16*xt inside the 48-col slab.
      const __bf16* pa =
          &A_lds[((r + dy) * CH + ctile * TX + ml) * TSTR + 16 * xt + 8 * mh];
      const v8bf alo = *(const v8bf*)pa;
      const v8bf ahi = *(const v8bf*)(pa + 16);
      // B fragment (16-bit 32x16): column ml, K runs [8mh, 8mh+7] and [16+8mh, +7].
      const __bf16* pb = &B_lds[((dy * 2 + xt) * TX + ml) * BSTR + 8 * mh];
      const v8bf blo = *(const v8bf*)pb;
      const v8bf bhi = *(const v8bf*)(pb + 16);

      v16bf av, bv;
#pragma unroll
      for (int i = 0; i < 8; ++i) {
        av[i] = alo[i]; av[i + 8] = ahi[i];
        bv[i] = blo[i]; bv[i + 8] = bhi[i];
      }
      acc = __builtin_amdgcn_wmma_f32_16x16x32_bf16(
          /*neg_a=*/false, av, /*neg_b=*/false, bv,
          /*c_mod=*/(short)0, acc, /*reuse_a=*/false, /*reuse_b=*/false);
    }

    // ---- Store: f32 C/D layout -> VGPR i is (M = i + 8*mh, N = ml);
    // lanes write contiguous 64B x-runs.
#pragma unroll
    for (int i = 0; i < 8; ++i) {
      const int c = ctile * TX + 8 * mh + i;
      out[(((size_t)b * CH + c) * H + y) * W + (xg + xt * TX + ml)] = acc[i];
    }
  }
}

extern "C" void kernel_launch(void* const* d_in, const int* in_sizes, int n_in,
                              void* d_out, int out_size, void* d_ws, size_t ws_size,
                              hipStream_t stream) {
  (void)in_sizes; (void)n_in; (void)out_size; (void)d_ws; (void)ws_size;
  const float* aff = (const float*)d_in[0];   // [4,169,192,192] f32
  const float* in2 = (const float*)d_in[1];   // [4,64,192,192]  f32
  float* out = (float*)d_out;                 // [4,64,192,192]  f32

  dim3 grid(W / XB, H / YB, BATCH);           // 6 x 48 x 4 workgroups
  mx_assemble_wmma<<<grid, dim3(NTHR), 0, stream>>>(aff, in2, out);
}
